// SpKBGATModified_71236327571612
// MI455X (gfx1250) — matrix operand order repaired and checked by reading, not compiled, not run on hardware
//
#include <hip/hip_runtime.h>
#include <hip/hip_bf16.h>
#include <math.h>

// ---------------------------------------------------------------------------
// KBGAT (SpKBGATModified) forward for MI455X / gfx1250.
//  * dense projections: bf16 WMMA (v_wmma_f32_16x16x32_bf16), operands packed
//    to padded bf16 so the K-loop is guard-free: 2 x 32B loads + 1 wmma / iter
//  * edge phase: per-node projected tables + gather + global_atomic_add_f32
// Workspace: ~234 MB.
// ---------------------------------------------------------------------------

typedef __attribute__((ext_vector_type(16))) __bf16 v16bf;
typedef __attribute__((ext_vector_type(8)))  float  v8f;

#define N_ENT_C 30000
#define N_REL_C 500
#define D_IN_C  200
#define D_HID_C 400
#define E1_C    250000
#define E2_C    50000
#define ETOT_C  300000
#define NB_C    8192
#define ALPHA_C 0.2f

__device__ __forceinline__ void atomAddF(float* p, float v) {
  unsafeAtomicAdd(p, v);   // -> global_atomic_add_f32
}

// A-operand k-permutation within each 32-k block (swap octets 1<->2) so the
// 16x32 bf16 A fragment (lane half h: k in {8h..8h+7} u {16+8h..16+8h+7})
// becomes one contiguous 32-byte load at offset kb + 16*h.
__device__ __forceinline__ int aperm(int d) {
  int t = d & 31, o = t >> 3;
  o = (o == 1) ? 2 : ((o == 2) ? 1 : o);
  return (d & ~31) | (o << 3) | (t & 7);
}

// ---------------------------------------------------------------------------
// WMMA GEMM on pre-packed operands.
//   A: bf16 [Mp, Kp], k-permuted rows (aperm), zero padded
//   B: bf16 [Np, Kp] = B^T col-major, k-permuted (aperm), zero padded
//   C: f32  [M, N] (ldc), store bounds-guarded only
// 1 wave per 16x16 C tile, 4 M-tiles per 128-thread block.
// ---------------------------------------------------------------------------
__global__ __launch_bounds__(128)
void gemm_bf16_wmma(const __bf16* __restrict__ A, const __bf16* __restrict__ B,
                    float* __restrict__ C, int ldc, int M, int N, int Kp)
{
  const int wave  = threadIdx.x >> 5;
  const int lane  = threadIdx.x & 31;
  const int mTile = (blockIdx.x * 4 + wave) * 16;
  const int nTile = blockIdx.y * 16;
  if (mTile >= M) return;

  const int mn   = lane & 15;
  const int half = lane >> 4;
  const __bf16* Ab = A + (size_t)(mTile + mn) * Kp + 16 * half;
  const __bf16* Bb = B + (size_t)(nTile + mn) * Kp + 16 * half;

  v8f acc = {0.f, 0.f, 0.f, 0.f, 0.f, 0.f, 0.f, 0.f};
  for (int kb = 0; kb < Kp; kb += 32) {
    v16bf a = *(const v16bf*)(Ab + kb);
    v16bf b = *(const v16bf*)(Bb + kb);
    acc = __builtin_amdgcn_wmma_f32_16x16x32_bf16(false, a, false, b,
                                                  (short)0, acc, false, false);
  }
#pragma unroll
  for (int r = 0; r < 8; ++r) {
    int m = mTile + 8 * half + r;
    int n = nTile + mn;
    if (m < M && n < N) C[(size_t)m * ldc + n] = acc[r];
  }
}

// ---------------------------------------------------------------------------
// l2-normalize rows of f32 src and emit packed bf16 A operand (zero padded).
// Wave per (padded) row.
// ---------------------------------------------------------------------------
__global__ void l2norm_pack_bf16(const float* __restrict__ X, __bf16* __restrict__ Y,
                                 int rows, int rowsPad, int D, int Kp)
{
  int row  = (int)((blockIdx.x * blockDim.x + threadIdx.x) >> 5);
  int lane = threadIdx.x & 31;
  if (row >= rowsPad) return;
  __bf16* y = Y + (size_t)row * Kp;
  if (row >= rows) {
    for (int d = lane; d < Kp; d += 32) y[d] = (__bf16)0.f;
    return;
  }
  const float* x = X + (size_t)row * D;
  float s = 0.f;
  for (int d = lane; d < D; d += 32) { float v = x[d]; s += v * v; }
#pragma unroll
  for (int off = 16; off > 0; off >>= 1) s += __shfl_xor(s, off, 32);
  float inv = 1.f / fmaxf(sqrtf(s), 1e-12f);
  for (int d = lane; d < Kp; d += 32) {
    float v = (d < D) ? x[d] * inv : 0.f;
    y[aperm(d)] = (__bf16)v;
  }
}

// plain f32 row-wise l2 normalize (for final outputs); Y may alias X
__global__ void l2norm_rows(const float* __restrict__ X, float* __restrict__ Y,
                            int rows, int D)
{
  int row  = (int)((blockIdx.x * blockDim.x + threadIdx.x) >> 5);
  int lane = threadIdx.x & 31;
  if (row >= rows) return;
  const float* x = X + (size_t)row * D;
  float s = 0.f;
  for (int d = lane; d < D; d += 32) { float v = x[d]; s += v * v; }
#pragma unroll
  for (int off = 16; off > 0; off >>= 1) s += __shfl_xor(s, off, 32);
  float inv = 1.f / fmaxf(sqrtf(s), 1e-12f);
  float* y = Y + (size_t)row * D;
  for (int d = lane; d < D; d += 32) y[d] = x[d] * inv;
}

// pack f32 [M,K] -> k-permuted, zero-padded bf16 A [Mp,Kp]
__global__ void pack_a_bf16(const float* __restrict__ src, __bf16* __restrict__ dst,
                            int M, int Mp, int K, int Kp)
{
  size_t idx = (size_t)blockIdx.x * blockDim.x + threadIdx.x;
  if (idx >= (size_t)Mp * Kp) return;
  int r = (int)(idx / Kp), k = (int)(idx % Kp);
  float v = (r < M && k < K) ? src[(size_t)r * K + k] : 0.f;
  dst[(size_t)r * Kp + aperm(k)] = (__bf16)v;
}

// pack B operand -> bf16 col-major [Np,Kp], zero padded, k-permuted (aperm).
//   transB=1: B(k,n) = B1[n*ldb+k] (+ B2[n*ldb+k])  (slice of a / a_out, .T)
//   transB=0: B(k,n) = B1[k*ldb+n]                  (plain weight matrix)
__global__ void pack_b_bf16(const float* __restrict__ B1, const float* __restrict__ B2,
                            int ldb, int transB, __bf16* __restrict__ dst,
                            int N, int Np, int K, int Kp)
{
  size_t idx = (size_t)blockIdx.x * blockDim.x + threadIdx.x;
  if (idx >= (size_t)Np * Kp) return;
  int n = (int)(idx / Kp), k = (int)(idx % Kp);
  float v = 0.f;
  if (n < N && k < K) {
    size_t s = transB ? ((size_t)n * ldb + k) : ((size_t)k * ldb + n);
    v = B1[s];
    if (B2) v += B2[s];
  }
  dst[(size_t)n * Kp + aperm(k)] = (__bf16)v;
}

// s[row] = dot(P[row,:], v)   (wave per row)
__global__ void rows_dot(const float* __restrict__ P, const float* __restrict__ v,
                         float* __restrict__ s, int rows, int D)
{
  int row  = (int)((blockIdx.x * blockDim.x + threadIdx.x) >> 5);
  int lane = threadIdx.x & 31;
  if (row >= rows) return;
  const float* p = P + (size_t)row * D;
  float acc = 0.f;
  for (int d = lane; d < D; d += 32) acc += p[d] * v[d];
#pragma unroll
  for (int off = 16; off > 0; off >>= 1) acc += __shfl_xor(acc, off, 32);
  if (lane == 0) s[row] = acc;
}

// ---------------------------------------------------------------------------
// Edge gather/scatter (wave per edge); see round-0 derivation:
//   m[d] = tabA[n0,d]+tabB[n1,d]+tabR[r0,d] (+tabR[r1,d]+tabRE[n0,d]+tabRE[n1,d] nhop)
//   w    = exp(-leaky_relu(s, 0.2)), s from precomputed per-row dot tables
//   rowsum[n0] += w ; hacc[n0,:] += w * m
// ---------------------------------------------------------------------------
__global__ void edge_scatter(const int* __restrict__ eL0, const int* __restrict__ eL1,
                             const int* __restrict__ etype, const int* __restrict__ tin,
                             const float* __restrict__ tabA, const float* __restrict__ tabB,
                             const float* __restrict__ tabR, const float* __restrict__ tabRE,
                             const float* __restrict__ sA, const float* __restrict__ sB,
                             const float* __restrict__ sR, const float* __restrict__ sRE,
                             float* __restrict__ rowsum, float* __restrict__ hacc, int D)
{
  int e    = (int)(((size_t)blockIdx.x * blockDim.x + threadIdx.x) >> 5);
  int lane = threadIdx.x & 31;
  if (e >= ETOT_C) return;

  int n0, n1, r0, r1 = -1;
  if (e < E1_C) {
    n0 = eL0[e]; n1 = eL1[e]; r0 = etype[e];
  } else {
    int i = e - E1_C;                      // nhop row: [e1, t0, t1, e0]
    n0 = tin[i * 4 + 3]; n1 = tin[i * 4 + 0];
    r0 = tin[i * 4 + 1]; r1 = tin[i * 4 + 2];
  }
  bool nh = (r1 >= 0);

  float s = sA[n0] + sB[n1] + sR[r0];
  if (nh) { s += sR[r1]; if (sRE) s += sRE[n0] + sRE[n1]; }
  float lr = (s > 0.f) ? s : ALPHA_C * s;
  float w  = __expf(-lr);
  if (lane == 0) atomAddF(rowsum + n0, w);

  const float* pA  = tabA + (size_t)n0 * D;
  const float* pB  = tabB + (size_t)n1 * D;
  const float* pR0 = tabR + (size_t)r0 * D;
  const float* pR1 = nh ? (tabR + (size_t)r1 * D) : nullptr;
  const float* pE0 = (nh && tabRE) ? (tabRE + (size_t)n0 * D) : nullptr;
  const float* pE1 = (nh && tabRE) ? (tabRE + (size_t)n1 * D) : nullptr;
  float* out = hacc + (size_t)n0 * D;

  for (int d = lane; d < D; d += 32) {
    float m = pA[d] + pB[d] + pR0[d];
    if (pR1) m += pR1[d];
    if (pE0) m += pE0[d] + pE1[d];
    atomAddF(out + d, w * m);
  }
}

// layer-1: out1A[n, aperm(coloff+d)] = bf16(elu(hacc/rowsum))   (packed A operand)
__global__ void finalize_elu_pack(const float* __restrict__ hacc, const float* __restrict__ rowsum,
                                  __bf16* __restrict__ dst, int Kp, int coloff, int D, int rows)
{
  size_t idx = (size_t)blockIdx.x * blockDim.x + threadIdx.x;
  if (idx >= (size_t)rows * D) return;
  int n = (int)(idx / D), d = (int)(idx % D);
  float rs = rowsum[n];
  rs = (rs == 0.f) ? 1e-12f : rs;
  float h = hacc[idx] / rs;
  float o = (h > 0.f) ? h : (__expf(h) - 1.f);
  dst[(size_t)n * Kp + aperm(coloff + d)] = (__bf16)o;
}

// layer-2: dst[n,d] = elu(hacc/rowsum), f32 (dst may alias hacc)
__global__ void finalize_elu(const float* __restrict__ hacc, const float* __restrict__ rowsum,
                             float* __restrict__ dst, int D, int rows)
{
  size_t idx = (size_t)blockIdx.x * blockDim.x + threadIdx.x;
  if (idx >= (size_t)rows * D) return;
  int n = (int)(idx / D);
  float rs = rowsum[n];
  rs = (rs == 0.f) ? 1e-12f : rs;
  float h = hacc[idx] / rs;
  dst[idx] = (h > 0.f) ? h : (__expf(h) - 1.f);
}

__global__ void mask_scatter(const int* __restrict__ batch, float* __restrict__ mask)
{
  int i = blockIdx.x * blockDim.x + threadIdx.x;
  if (i < NB_C) mask[batch[i * 3 + 2]] = 1.0f;
}

__global__ void add_masked(float* __restrict__ C, const float* __restrict__ mask,
                           const float* __restrict__ out2)
{
  size_t idx = (size_t)blockIdx.x * blockDim.x + threadIdx.x;
  if (idx >= (size_t)N_ENT_C * D_HID_C) return;
  int n = (int)(idx / D_HID_C);
  C[idx] += mask[n] * out2[idx];
}

// ---------------------------------------------------------------------------
extern "C" void kernel_launch(void* const* d_in, const int* in_sizes, int n_in,
                              void* d_out, int out_size, void* d_ws, size_t ws_size,
                              hipStream_t stream)
{
  (void)in_sizes; (void)n_in; (void)out_size; (void)ws_size;

  const float* entity_emb   = (const float*)d_in[0];
  const float* relation_emb = (const float*)d_in[1];
  const float* a_heads      = (const float*)d_in[2];   // [2,200,600]
  const float* a2_heads     = (const float*)d_in[3];   // [2,1,200]
  const float* W_rel        = (const float*)d_in[4];   // [200,400]
  const float* a_out        = (const float*)d_in[5];   // [400,1200]
  const float* a2_out       = (const float*)d_in[6];   // [1,400]
  const float* W_entities   = (const float*)d_in[7];   // [200,400]
  const int*   edge_list    = (const int*)d_in[8];     // [2,E1]
  const int*   edge_type    = (const int*)d_in[9];
  const int*   tin          = (const int*)d_in[10];    // [E2,4]
  const int*   batch        = (const int*)d_in[11];    // [NB,3]

  float* out_ent_o = (float*)d_out;
  float* out_rel_o = (float*)d_out + (size_t)N_ENT_C * D_HID_C;

  const int KP1 = 224;   // pad 200 -> 224 (mult of 32)
  const int KP2 = 416;   // pad 400 -> 416
  const int RELP = 512;  // pad 500 -> 512 rows

  // ----- workspace carving (bytes, 256-aligned chunks) -----
  char* base = (char*)d_ws;
  size_t off = 0;
  auto alloc = [&](size_t bytes) -> char* {
    char* p = base + off;
    off = (off + bytes + 255) & ~(size_t)255;
    return p;
  };
  __bf16* entA  = (__bf16*)alloc((size_t)N_ENT_C * KP1 * 2);  // packed l2norm(ent)
  __bf16* relA  = (__bf16*)alloc((size_t)RELP * KP1 * 2);     // packed l2norm(rel)
  __bf16* out1A = (__bf16*)alloc((size_t)N_ENT_C * KP2 * 2);  // packed layer-1 out
  __bf16* orelA = (__bf16*)alloc((size_t)RELP * KP2 * 2);     // packed out_rel
  float*  ORELf = (float*)alloc((size_t)N_REL_C * D_HID_C * 4);
  __bf16* Bp    = (__bf16*)alloc((size_t)416 * 416 * 2);      // shared packed B
  float*  arena = (float*)alloc((size_t)48400000 * 4);        // phase B/D union
  // phase B (per head, sequential reuse)
  float* GI  = arena;                 float* GJ  = arena + 6000000;
  float* GR  = arena + 12000000;      float* HAC = arena + 12100000;
  float* SGI = arena + 18100000;      float* SGJ = arena + 18130000;
  float* SGR = arena + 18160000;      float* RS  = arena + 18160512;
  // phase D (reuses arena)
  float* PI2  = arena;                float* PJ2  = arena + 12000000;
  float* PR2  = arena + 24000000;     float* TBL  = arena + 36000000;
  float* HAC2 = arena + 36200000;     float* SPI  = arena + 48200000;
  float* SPJ  = arena + 48230000;     float* SPR  = arena + 48260000;
  float* STB  = arena + 48290000;     float* RS2  = arena + 48290512;
  float* MASK = arena + 48320512;

  const int* e0 = edge_list;
  const int* e1 = edge_list + E1_C;

  auto cdiv = [](size_t a, size_t b) { return (unsigned)((a + b - 1) / b); };
  auto wave_rows = [](int rows) { return dim3((rows * 32 + 255) / 256); };

  // pack B then run WMMA GEMM: C[M,N] = A(packed) * B
  auto gemm = [&](const __bf16* A, const float* B1, const float* B2, int ldb,
                  int transB, float* C, int ldc, int M, int N, int K, int Kp) {
    int Np = (N + 15) & ~15;
    pack_b_bf16<<<cdiv((size_t)Np * Kp, 256), 256, 0, stream>>>(
        B1, B2, ldb, transB, Bp, N, Np, K, Kp);
    dim3 grid((M + 63) / 64, (N + 15) / 16);
    gemm_bf16_wmma<<<grid, 128, 0, stream>>>(A, Bp, C, ldc, M, N, Kp);
  };

  const int EDGE_BLOCKS = (int)(((size_t)ETOT_C * 32 + 255) / 256);

  // 1) l2-normalize embeddings straight into packed bf16 A operands
  l2norm_pack_bf16<<<wave_rows(N_ENT_C), 256, 0, stream>>>(
      entity_emb, entA, N_ENT_C, N_ENT_C, D_IN_C, KP1);
  l2norm_pack_bf16<<<wave_rows(RELP), 256, 0, stream>>>(
      relation_emb, relA, N_REL_C, RELP, D_IN_C, KP1);

  hipMemsetAsync(out1A, 0, (size_t)N_ENT_C * KP2 * 2, stream);  // covers pad cols

  // 2) layer-1 attention heads
  for (int h = 0; h < 2; ++h) {
    const float* ah  = a_heads + (size_t)h * D_IN_C * 600;   // [200,600]
    const float* a2h = a2_heads + (size_t)h * D_IN_C;

    gemm(entA, ah,       ah + 400, 600, 1, GI, D_IN_C, N_ENT_C, D_IN_C, D_IN_C, KP1); // Gi=ent@(ai+ar)^T
    gemm(entA, ah + 200, ah + 400, 600, 1, GJ, D_IN_C, N_ENT_C, D_IN_C, D_IN_C, KP1); // Gj=ent@(aj+ar)^T
    gemm(relA, ah + 400, nullptr,  600, 1, GR, D_IN_C, N_REL_C, D_IN_C, D_IN_C, KP1); // Gr=rel@ar^T

    rows_dot<<<wave_rows(N_ENT_C), 256, 0, stream>>>(GI, a2h, SGI, N_ENT_C, D_IN_C);
    rows_dot<<<wave_rows(N_ENT_C), 256, 0, stream>>>(GJ, a2h, SGJ, N_ENT_C, D_IN_C);
    rows_dot<<<wave_rows(N_REL_C), 256, 0, stream>>>(GR, a2h, SGR, N_REL_C, D_IN_C);

    hipMemsetAsync(HAC, 0, (size_t)N_ENT_C * D_IN_C * 4, stream);
    hipMemsetAsync(RS,  0, (size_t)N_ENT_C * 4, stream);

    edge_scatter<<<EDGE_BLOCKS, 256, 0, stream>>>(e0, e1, edge_type, tin,
        GI, GJ, GR, nullptr, SGI, SGJ, SGR, nullptr, RS, HAC, D_IN_C);

    finalize_elu_pack<<<cdiv((size_t)N_ENT_C * D_IN_C, 256), 256, 0, stream>>>(
        HAC, RS, out1A, KP2, h * D_IN_C, D_IN_C, N_ENT_C);
  }

  // 3) out_rel = rel @ W_rel, then pack as A operand
  gemm(relA, W_rel, nullptr, D_HID_C, 0, ORELf, D_HID_C, N_REL_C, D_HID_C, D_IN_C, KP1);
  pack_a_bf16<<<cdiv((size_t)RELP * KP2, 256), 256, 0, stream>>>(
      ORELf, orelA, N_REL_C, RELP, D_HID_C, KP2);

  // 4) layer-2 node/relation tables
  gemm(out1A, a_out + 0,   nullptr, 1200, 1, PI2, D_HID_C, N_ENT_C, D_HID_C, D_HID_C, KP2);
  gemm(out1A, a_out + 400, nullptr, 1200, 1, PJ2, D_HID_C, N_ENT_C, D_HID_C, D_HID_C, KP2);
  gemm(out1A, a_out + 800, nullptr, 1200, 1, PR2, D_HID_C, N_ENT_C, D_HID_C, D_HID_C, KP2);
  gemm(orelA, a_out + 800, nullptr, 1200, 1, TBL, D_HID_C, N_REL_C, D_HID_C, D_HID_C, KP2);

  rows_dot<<<wave_rows(N_ENT_C), 256, 0, stream>>>(PI2, a2_out, SPI, N_ENT_C, D_HID_C);
  rows_dot<<<wave_rows(N_ENT_C), 256, 0, stream>>>(PJ2, a2_out, SPJ, N_ENT_C, D_HID_C);
  rows_dot<<<wave_rows(N_ENT_C), 256, 0, stream>>>(PR2, a2_out, SPR, N_ENT_C, D_HID_C);
  rows_dot<<<wave_rows(N_REL_C), 256, 0, stream>>>(TBL, a2_out, STB, N_REL_C, D_HID_C);

  hipMemsetAsync(HAC2, 0, (size_t)N_ENT_C * D_HID_C * 4, stream);
  hipMemsetAsync(RS2,  0, (size_t)N_ENT_C * 4, stream);

  edge_scatter<<<EDGE_BLOCKS, 256, 0, stream>>>(e0, e1, edge_type, tin,
      PI2, PJ2, TBL, PR2, SPI, SPJ, STB, SPR, RS2, HAC2, D_HID_C);

  finalize_elu<<<cdiv((size_t)N_ENT_C * D_HID_C, 256), 256, 0, stream>>>(
      HAC2, RS2, HAC2, D_HID_C, N_ENT_C);   // out_ent2, in place

  // 5) mask + final projections + l2norm
  hipMemsetAsync(MASK, 0, (size_t)N_ENT_C * 4, stream);
  mask_scatter<<<(NB_C + 255) / 256, 256, 0, stream>>>(batch, MASK);

  gemm(entA, W_entities, nullptr, D_HID_C, 0, out_ent_o, D_HID_C, N_ENT_C, D_HID_C, D_IN_C, KP1);
  add_masked<<<cdiv((size_t)N_ENT_C * D_HID_C, 256), 256, 0, stream>>>(out_ent_o, MASK, HAC2);
  l2norm_rows<<<wave_rows(N_ENT_C), 256, 0, stream>>>(out_ent_o, out_ent_o, N_ENT_C, D_HID_C);

  gemm(relA, W_entities, nullptr, D_HID_C, 0, out_rel_o, D_HID_C, N_REL_C, D_HID_C, D_IN_C, KP1);
  l2norm_rows<<<wave_rows(N_REL_C), 256, 0, stream>>>(out_rel_o, out_rel_o, N_REL_C, D_HID_C);
}
